// AttentionLayer_30288109371738
// MI455X (gfx1250) — matrix operand order repaired and checked by reading, not compiled, not run on hardware
//
#include <hip/hip_runtime.h>
#include <hip/hip_bf16.h>
#include <math.h>

// ---- problem constants (B=32, S=128 fixed by the reference) ----
#define BS      4096      // B*S tokens
#define WLEN    20        // words per token
#define DIM     300       // D1 == D2 == OUT
#define NPAD    304       // 19 * 16
#define KPAD    320       // 10 * 32
#define NTILES  19
#define KSTEPS  10
#define NEG_INF (-1.0e9f)
#define WT_HALVES   (NPAD * KPAD)     // halves per (padded, transposed) weight matrix
#define VECB_OFFSET (393216)          // 384 KB: bf16 vector region in d_ws
#define VP_OFFSET   (3014656)         // vecpart f32 region in d_ws (~8 MB total ws use)

typedef __attribute__((ext_vector_type(16))) unsigned short v16u;
typedef __attribute__((ext_vector_type(8)))  unsigned short v8u;
typedef __attribute__((ext_vector_type(16))) __bf16        v16bf;
typedef __attribute__((ext_vector_type(8)))  float         v8f;

// f32 -> bf16 round-to-nearest-even (bit pattern)
static __device__ __forceinline__ unsigned short f2bf(float f) {
    unsigned u = __builtin_bit_cast(unsigned, f);
    unsigned r = u + 0x7FFFu + ((u >> 16) & 1u);
    return (unsigned short)(r >> 16);
}

// gfx1250 has a hardware tanh transcendental (ISA: TRANS32 ops incl. TANH)
static __device__ __forceinline__ float fast_tanh(float x) {
#if __has_builtin(__builtin_amdgcn_tanhf)
    return __builtin_amdgcn_tanhf(x);
#elif __has_builtin(__builtin_amdgcn_tanh_f32)
    return __builtin_amdgcn_tanh_f32(x);
#else
    return tanhf(x);
#endif
}

static __device__ __forceinline__ v8f wmma_bf16(v16u a, v16u b, v8f c) {
    return __builtin_amdgcn_wmma_f32_16x16x32_bf16(
        false, __builtin_bit_cast(v16bf, a),
        false, __builtin_bit_cast(v16bf, b),
        (short)0, c, false, false);
}

// =====================================================================
// Kernel 0a: one-time weight prep.  WT[p][n][k] = bf16(W[n][p*300+k]),
// zero-padded n<304, k<320, n-major so a WMMA B-fragment is two
// contiguous 16B loads per lane.  p=0: Ws, p=1: Wv.   (380 KB)
// =====================================================================
__global__ __launch_bounds__(256)
void prepW_kernel(const float* __restrict__ W,        // [300][600]
                  unsigned int* __restrict__ wt)      // [2][304][160] packed bf16 pairs
{
    const int i  = blockIdx.x * 256 + threadIdx.x;    // 380*256 = 97280 = 2*304*160 exact
    const int k2 = i % (KPAD / 2);
    const int n  = (i / (KPAD / 2)) % NPAD;
    const int p  = i / ((KPAD / 2) * NPAD);
    const int k  = k2 * 2;
    unsigned lo = 0, hi = 0;
    if (n < DIM) {
        const float* wp = W + n * (2 * DIM) + p * DIM;
        if (k     < DIM) lo = f2bf(wp[k]);
        if (k + 1 < DIM) hi = f2bf(wp[k + 1]);
    }
    wt[i] = lo | (hi << 16);
}

// =====================================================================
// Kernel 0b: vector prep.  vecb[row][k] = bf16(vector[row][k]), k<320 pad.
// =====================================================================
__global__ __launch_bounds__(256)
void prepV_kernel(const float* __restrict__ vec,      // [4096][300]
                  unsigned int* __restrict__ vecb)    // [4096][160] packed bf16 pairs
{
    const int i   = blockIdx.x * 256 + threadIdx.x;   // 2560*256 = 655360 = 4096*160 exact
    const int k2  = i % (KPAD / 2);
    const int row = i / (KPAD / 2);
    const int k   = k2 * 2;
    const float* rp = vec + row * DIM;
    unsigned lo = (k     < DIM) ? f2bf(rp[k])     : 0u;
    unsigned hi = (k + 1 < DIM) ? f2bf(rp[k + 1]) : 0u;
    vecb[i] = lo | (hi << 16);
}

// =====================================================================
// Kernel 1: vecpart[bs][n] = vector[bs,:] . Wv[n,:] + b[n]
// GEMM  M=4096  K=300(->320)  N=300(->304), 1 wave per 16x16 tile.
// =====================================================================
__global__ __launch_bounds__(256)
void vecpart_kernel(const unsigned short* __restrict__ vecb, // [4096][320] bf16
                    const unsigned short* __restrict__ wt,   // WT (p=0 Ws, p=1 Wv)
                    const float* __restrict__ bias,          // [300]
                    float* __restrict__ vp)                  // [4096][304]
{
    const int lane = threadIdx.x & 31;
    const int wave = threadIdx.x >> 5;
    const int tile = blockIdx.x * 8 + wave;            // 608*8 = 4864 = 256*19 exact
    const int mt = tile / NTILES;
    const int nt = tile % NTILES;

    const int nl   = lane & 15;
    const int mrow = mt * 16 + nl;                     // A row (< 4096 always)
    const int ncol = nt * 16 + nl;                     // B/C column (< 304)
    const int kHiA = (lane >= 16) ? 8  : 0;            // ISA 16-bit A layout
    const int kHiB = (lane >= 16) ? 8  : 0;            // ISA 16-bit B layout
    const unsigned short* WvT = wt + WT_HALVES;        // p=1 matrix

    v8f acc = {};
    for (int kt = 0; kt < KSTEPS; ++kt) {
        const int kb = kt * 32;
        // A fragment: two contiguous 16B runs of prepped bf16 vector row
        union { v16u v; v8u h[2]; } ua;
        const unsigned short* ap = vecb + mrow * KPAD + kb + kHiA;
        ua.h[0] = *(const v8u*)ap;
        ua.h[1] = *(const v8u*)(ap + 16);
        // B fragment: two contiguous 16B runs of prepped bf16 WvT
        union { v16u v; v8u h[2]; } ub;
        const unsigned short* bp = WvT + ncol * KPAD + kb + kHiB;
        ub.h[0] = *(const v8u*)bp;
        ub.h[1] = *(const v8u*)(bp + 16);
        acc = wmma_bf16(ua.v, ub.v, acc);
    }

    const float bn = (ncol < DIM) ? bias[ncol] : 0.0f;
    #pragma unroll
    for (int r = 0; r < 8; ++r) {
        const int m = (lane < 16) ? r : (8 + r);       // ISA f32 C/D layout
        vp[(mt * 16 + m) * NPAD + ncol] = acc[r] + bn;
    }
}

// =====================================================================
// Kernel 2: per-token attention.  1 block = 1 (b,s) token, 8 waves.
// =====================================================================
__global__ __launch_bounds__(256)
void attn_kernel(const float* __restrict__ seq,       // [4096][20][300]
                 const int*   __restrict__ cmask,     // [4096][20]
                 const unsigned short* __restrict__ wt, // prepped WT (p=0: WsT)
                 const float* __restrict__ vvec,      // [300]
                 const float* __restrict__ vp,        // [4096][304]
                 float* __restrict__ out)             // [4096][300]
{
    __shared__ float          sSeq[WLEN * DIM];       // 24000 B, f32 tile for final pass
    __shared__ unsigned short sA[32 * KPAD];          // 20480 B, zero-padded bf16 A
    __shared__ float          sVp[NPAD];
    __shared__ float          sV[NPAD];
    __shared__ float          sScore[32];

    const int bs   = blockIdx.x;
    const int tid  = threadIdx.x;
    const int lane = tid & 31;
    const int wave = tid >> 5;

    // ---- Phase A: stage token tile into LDS (f32 copy + bf16 A matrix) ----
    const float4* g4 = (const float4*)(seq + (size_t)bs * (WLEN * DIM));
    for (int i = tid; i < (WLEN * DIM) / 4; i += 256) {
        float4 x = g4[i];
        ((float4*)sSeq)[i] = x;
        const int j = i * 4;                // DIM % 4 == 0 -> no row straddle
        const int m = j / DIM, k = j % DIM;
        unsigned short* p = &sA[m * KPAD + k];
        p[0] = f2bf(x.x); p[1] = f2bf(x.y); p[2] = f2bf(x.z); p[3] = f2bf(x.w);
    }
    // zero pad rows>=20 and k>=300 (uint granularity; DIM even -> no straddle)
    unsigned int* sA32 = (unsigned int*)sA;
    for (int i = tid; i < 32 * (KPAD / 2); i += 256) {
        const int m = i / (KPAD / 2), kk = i % (KPAD / 2);
        if (m >= WLEN || kk >= DIM / 2) sA32[i] = 0;
    }
    for (int i = tid; i < NPAD; i += 256) {
        sVp[i] = vp[bs * NPAD + i];                   // pad cols written as 0 by k1
        sV[i]  = (i < DIM) ? vvec[i] : 0.0f;
    }
    if (tid < 32) sScore[tid] = 0.0f;
    __syncthreads();

    // ---- Phase B: lin = A(32x320) @ WsT(320x304), tanh, score reduce ----
    const int nslots = (wave < 3) ? 3 : 2;            // nt = wave, wave+8, wave+16(<19)
    v8f acc0[3] = {v8f{}, v8f{}, v8f{}};              // rows 0..15
    v8f acc1[3] = {v8f{}, v8f{}, v8f{}};              // rows 16..31 (only 16..19 live)
    const int nl   = lane & 15;
    const int kHiA = (lane >= 16) ? 8  : 0;
    const int kHiB = (lane >= 16) ? 8  : 0;

    for (int kt = 0; kt < KSTEPS; ++kt) {
        const int kb = kt * 32;
        const int c0 = kb + kHiA;
        // A fragments from LDS: two ds_load_b128 per row per lane
        union { v16u v; v8u h[2]; } ua0, ua1;
        ua0.h[0] = *(const v8u*)&sA[ nl       * KPAD + c0];
        ua0.h[1] = *(const v8u*)&sA[ nl       * KPAD + c0 + 16];
        ua1.h[0] = *(const v8u*)&sA[(16 + nl) * KPAD + c0];
        ua1.h[1] = *(const v8u*)&sA[(16 + nl) * KPAD + c0 + 16];

        for (int s = 0; s < nslots; ++s) {
            const int nt = wave + 8 * s;
            const int n  = nt * 16 + nl;
            // B fragment: two contiguous 16B runs of prepped bf16 WsT (L2-hot)
            union { v16u v; v8u h[2]; } ub;
            const unsigned short* bp = wt + n * KPAD + kb + kHiB;
            ub.h[0] = *(const v8u*)bp;
            ub.h[1] = *(const v8u*)(bp + 16);
            __builtin_prefetch(bp + 32, 0, 1);        // next K-step row -> global_prefetch
            acc0[s] = wmma_bf16(ua0.v, ub.v, acc0[s]);
            acc1[s] = wmma_bf16(ua1.v, ub.v, acc1[s]);
        }
    }

    // epilogue: score[w] += tanh(lin[w][n]) * v[n]  (LDS float atomics, HW tanh)
    for (int s = 0; s < nslots; ++s) {
        const int nt = wave + 8 * s;
        const int n  = nt * 16 + nl;
        const float vpn = sVp[n];
        const float vwn = sV[n];                      // 0 for pad cols
        #pragma unroll
        for (int r = 0; r < 8; ++r) {
            const int m = (lane < 16) ? r : (8 + r);  // < 16 < WLEN always
            float t0 = fast_tanh(acc0[s][r] + vpn);
            atomicAdd(&sScore[m], t0 * vwn);
            const int m1 = 16 + m;
            if (m1 < WLEN) {
                float t1 = fast_tanh(acc1[s][r] + vpn);
                atomicAdd(&sScore[m1], t1 * vwn);
            }
        }
    }
    __syncthreads();

    // ---- Phase C: masked softmax over words + alpha-weighted sum ----
    const int* cm = cmask + bs * WLEN;
    float al[WLEN];
    float mx = -3.0e38f;
    #pragma unroll
    for (int w = 0; w < WLEN; ++w) {
        float sc = (cm[w] == 0) ? NEG_INF : sScore[w];
        al[w] = sc;
        mx = fmaxf(mx, sc);
    }
    float den = 0.0f;
    #pragma unroll
    for (int w = 0; w < WLEN; ++w) { float e = __expf(al[w] - mx); al[w] = e; den += e; }
    const float inv = 1.0f / den;

    for (int d = tid; d < DIM; d += 256) {
        float o = 0.0f;
        #pragma unroll
        for (int w = 0; w < WLEN; ++w) o += al[w] * sSeq[w * DIM + d];
        out[(size_t)bs * DIM + d] = o * inv;
    }
}

// =====================================================================
extern "C" void kernel_launch(void* const* d_in, const int* in_sizes, int n_in,
                              void* d_out, int out_size, void* d_ws, size_t ws_size,
                              hipStream_t stream) {
    const float* seq   = (const float*)d_in[0];   // [32,128,20,300]
    const float* vec   = (const float*)d_in[1];   // [32,128,300]
    const int*   cmask = (const int*)  d_in[2];   // [32,128,20]
    const float* W     = (const float*)d_in[3];   // [300,600]
    const float* bias  = (const float*)d_in[4];   // [300]
    const float* vvec  = (const float*)d_in[5];   // [300]
    float*       out   = (float*)d_out;           // [32,128,300]

    unsigned short* wtb  = (unsigned short*)d_ws;                       // 380 KB bf16 WT
    unsigned short* vecb = (unsigned short*)((char*)d_ws + VECB_OFFSET); // 2.6 MB bf16 vec
    float*          vpw  = (float*)((char*)d_ws + VP_OFFSET);           // [4096][304] f32

    prepW_kernel<<<380, 256, 0, stream>>>(W, (unsigned int*)wtb);
    prepV_kernel<<<2560, 256, 0, stream>>>(vec, (unsigned int*)vecb);
    vecpart_kernel<<<608, 256, 0, stream>>>(vecb, wtb, bias, vpw);
    attn_kernel<<<BS, 256, 0, stream>>>(seq, cmask, wtb, vvec, vpw, out);
}
